// MSCRED_80788334838003
// MI455X (gfx1250) — matrix operand order repaired and checked by reading, not compiled, not run on hardware
//
#include <hip/hip_runtime.h>
#include <hip/hip_bf16.h>
#include <math.h>

typedef _Float16 v16h __attribute__((ext_vector_type(16)));
typedef _Float16 v2h  __attribute__((ext_vector_type(2)));
typedef float    v8f  __attribute__((ext_vector_type(8)));

#define SELU_SCALE 1.0507009873554805f
#define SELU_ALPHA 1.6732632423543772f

// ---------------------------------------------------------------------------
// Generic implicit-GEMM conv via WMMA f32 <- f16*f16, fp16-resident operands.
//   in  : f16 activations, NHWC, per-image stride iis (elements), offset ioff
//   wt  : f16 weights pre-transposed to [tap][Cout][Cin]
//   out : f32 (out32) and/or f16 (out16) NHWC
// spatial gather: ihn = ho*ostride + kh - pad ; valid iff ihn>=0,
//                 ihn % ldil == 0, ihn/ldil < Hin   (ldil>1 => transposed conv)
// One wave computes a 16(M) x 32(N) tile: two WMMA per 32-wide channel chunk,
// shared A fragment. Loop structure: taps (divisions hoisted) x Cin chunks
// (division-free, aligned 4-byte f16-pair gathers).
// WMMA stays in uniform control flow (EXEC all-1s): clamped decode, no branch.
// ---------------------------------------------------------------------------
__global__ void __launch_bounds__(32)
conv_wmma_kernel(const _Float16* __restrict__ in, const _Float16* __restrict__ wt,
                 const float* __restrict__ bias,
                 float* __restrict__ out32, _Float16* __restrict__ out16,
                 unsigned nimg, int Hin, int Win, int Cin,
                 int Hout, int Wout, int Cout,
                 int kh, int kw, int ostride, int ldil, int pad,
                 int iis, int ioff, int ois, int ooff,
                 int flip, int acc, int act)
{
  const int lane = threadIdx.x & 31;
  const int hv   = lane >> 4;      // wave half
  const int l16  = lane & 15;
  const unsigned m0 = blockIdx.x * 16u;
  const int  n0  = blockIdx.y * 32;
  const unsigned HWo  = (unsigned)(Hout * Wout);
  const unsigned Mtot = nimg * HWo;

  // wave-uniform decode of m0 (scalar division), cheap per-lane fixup
  const unsigned imgB = m0 / HWo;
  const unsigned remB = m0 - imgB * HWo;
  const bool arow = (m0 + (unsigned)l16) < Mtot;
  unsigned img = imgB, rem = remB + (unsigned)l16;     // l16 < 16 <= HWo
  if (rem >= HWo) { rem -= HWo; img++; }
  if (!arow)      { img = imgB; rem = remB; }          // clamp: valid location
  const int ho = (int)(rem / (unsigned)Wout);
  const int wo = (int)(rem - (unsigned)ho * (unsigned)Wout);

  const _Float16* inb = in + ioff + (long)img * iis;
  const int nn0 = n0 + l16;
  const int nn1 = nn0 + 16;
  const int ktaps = kh * kw;

  v8f c0, c1;
  #pragma unroll
  for (int i = 0; i < 8; ++i) { c0[i] = 0.f; c1[i] = 0.f; }

  for (int tap = 0; tap < ktaps; ++tap) {
    const int khh = tap / kw, kww = tap - khh * kw;
    const int ihn = ho * ostride + khh - pad;
    const int iwn = wo * ostride + kww - pad;
    bool sval = arow && ihn >= 0 && iwn >= 0;
    int ih = ihn, iw = iwn;
    if (ldil != 1) {
      sval = sval && ((ihn % ldil) == 0) && ((iwn % ldil) == 0);
      ih = ihn / ldil; iw = iwn / ldil;
    }
    sval = sval && ih < Hin && iw < Win;
    const _Float16* abase = inb + ((long)ih * Win + iw) * Cin;
    int wtap = tap;
    if (flip) wtap = (kh - 1 - khh) * kw + (kw - 1 - kww);
    const _Float16* wt0 = wt + (long)wtap * Cout * Cin;
    const _Float16* wr0 = wt0 + (long)nn0 * Cin;   // guarded by nn0 < Cout
    const _Float16* wr1 = wt0 + (long)nn1 * Cin;

    for (int cb = 0; cb < Cin; cb += 32) {
      if (sval && (cb + 32) < Cin)
        __builtin_prefetch(abase + cb + 32, 0, 1);   // global_prefetch_b8
      v16h af, b0f, b1f;
      #pragma unroll
      for (int v = 0; v < 8; ++v) {
        // A fragment: 16-bit A 16x32 layout; K pair (2v,2v+1):
        //   v<4 : K = 2v + p + 8*half ; v>=4 : K = 2v + 8 + p + 8*half
        const int kkA = (v < 4) ? (2*v + 8*hv) : (2*v + 8 + 8*hv);
        const int ciA = cb + kkA;
        _Float16 ax = (_Float16)0.f, ay = (_Float16)0.f;
        if (sval && ciA < Cin) {
          v2h t2 = *(const v2h*)(abase + ciA);       // aligned 4B load
          ax = t2[0]; ay = t2[1];
        }
        af[2*v]   = ax;
        af[2*v+1] = ay;
        // B fragment: same K order as A (v<4: 2v+p+8*half; v>=4: 2v+8+p+8*half)
        const int ciB = cb + kkA;
        _Float16 b00 = (_Float16)0.f, b01 = (_Float16)0.f;
        _Float16 b10 = (_Float16)0.f, b11 = (_Float16)0.f;
        if (ciB < Cin) {
          if (nn0 < Cout) { v2h t = *(const v2h*)(wr0 + ciB); b00 = t[0]; b01 = t[1]; }
          if (nn1 < Cout) { v2h t = *(const v2h*)(wr1 + ciB); b10 = t[0]; b11 = t[1]; }
        }
        b0f[2*v]   = b00;
        b0f[2*v+1] = b01;
        b1f[2*v]   = b10;
        b1f[2*v+1] = b11;
      }
      c0 = __builtin_amdgcn_wmma_f32_16x16x32_f16(false, af, false, b0f,
                                                  (short)0, c0, false, false);
      c1 = __builtin_amdgcn_wmma_f32_16x16x32_f16(false, af, false, b1f,
                                                  (short)0, c1, false, false);
    }
  }

  // f32 C/D layout: VGPR r -> M = r + 8*half, N = lane&15 (+16 second tile).
  // No division in epilogue: rem fixup like the prologue (off < 16 <= HWo).
  #pragma unroll
  for (int j = 0; j < 2; ++j) {
    const int nn = (j == 0) ? nn0 : nn1;
    #pragma unroll
    for (int rr = 0; rr < 8; ++rr) {
      const unsigned offm = (unsigned)(rr + 8*hv);
      if ((m0 + offm) < Mtot && nn < Cout) {
        unsigned img2 = imgB, rem2 = remB + offm;
        if (rem2 >= HWo) { rem2 -= HWo; img2++; }
        long oidx = (long)ooff + (long)img2 * ois + (long)rem2 * Cout + nn;
        float vv = (j == 0) ? c0[rr] : c1[rr];
        if (bias) vv += bias[nn];
        if (acc)  vv += out32[oidx];
        if (act == 1) vv = (vv > 0.f) ? SELU_SCALE * vv
                                      : SELU_SCALE * SELU_ALPHA * (__expf(vv) - 1.f);
        if (out32) out32[oidx] = vv;
        if (out16) out16[oidx] = (_Float16)vv;
      }
    }
  }
}

// ---------------------------------------------------------------------------
// weight convert f32 -> f16 with transpose [tap][Cin][Cout] -> [tap][Cout][Cin]
__global__ void wcvt_kernel(const float* __restrict__ w, _Float16* __restrict__ wt,
                            unsigned taps, unsigned Cin, unsigned Cout)
{
  unsigned n = blockIdx.x * blockDim.x + threadIdx.x;
  unsigned total = taps * Cin * Cout;
  if (n >= total) return;
  unsigned tap = n / (Cin * Cout);
  unsigned r = n - tap * Cin * Cout;
  unsigned ci = r / Cout;
  unsigned co = r - ci * Cout;
  wt[(tap * Cout + co) * Cin + ci] = (_Float16)w[n];
}

__global__ void f32_to_f16_kernel(const float* __restrict__ a, _Float16* __restrict__ o,
                                  unsigned n)
{
  unsigned i = blockIdx.x * blockDim.x + threadIdx.x;
  if (i < n) o[i] = (_Float16)a[i];
}

// ---------------------------------------------------------------------------
// ConvLSTM gates: z (f32) = [i|f|g|o]; writes c state (f32) and h (f16 into
// the (B,T,A,C) hs buffer slice for timestep t).
__global__ void lstm_gates_kernel(const float* __restrict__ z, float* __restrict__ cst,
                                  _Float16* __restrict__ hs,
                                  unsigned Bb, unsigned A, unsigned C, int t, int T,
                                  int first)
{
  unsigned n = blockIdx.x * blockDim.x + threadIdx.x;
  unsigned total = Bb * A * C;
  if (n >= total) return;
  unsigned b = n / (A * C);
  unsigned r = n - b * A * C;
  unsigned pos = r / C;
  unsigned ch = r - pos * C;
  const float* zb = z + ((long)b * A + pos) * (4 * C);
  float zi = zb[ch], zf = zb[C + ch], zg = zb[2 * C + ch], zo = zb[3 * C + ch];
  float cprev = first ? 0.f : cst[n];
  float hi = fminf(fmaxf(0.2f * zi + 0.5f, 0.f), 1.f);
  float hf = fminf(fmaxf(0.2f * zf + 0.5f, 0.f), 1.f);
  float ho = fminf(fmaxf(0.2f * zo + 0.5f, 0.f), 1.f);
  float cn = hf * cprev + hi * tanhf(zg);
  cst[n] = cn;
  hs[((long)b * T + t) * A * C + (long)pos * C + ch] = (_Float16)(ho * tanhf(cn));
}

// scores[t,b] = <hs[b,t], hs[b,T-1]> over A*C  (f16 in, f32 accumulate)
__global__ void att_scores_kernel(const _Float16* __restrict__ hs, float* __restrict__ scores,
                                  int Bb, int T, int AC)
{
  int tb = blockIdx.x;
  int t = tb / Bb, b = tb - t * Bb;
  const _Float16* xv = hs + ((long)b * T + t) * AC;
  const _Float16* yv = hs + ((long)b * T + (T - 1)) * AC;
  float s = 0.f;
  for (int i = threadIdx.x; i < AC; i += blockDim.x)
    s += (float)xv[i] * (float)yv[i];
  __shared__ float red[256];
  red[threadIdx.x] = s;
  __syncthreads();
  for (int off = 128; off > 0; off >>= 1) {
    if ((int)threadIdx.x < off) red[threadIdx.x] += red[threadIdx.x + off];
    __syncthreads();
  }
  if (threadIdx.x == 0) scores[(long)t * Bb + b] = red[0];
}

__global__ void att_softmax_kernel(const float* __restrict__ scores, float* __restrict__ att,
                                   int Bb, int T)
{
  int b = blockIdx.x * blockDim.x + threadIdx.x;
  if (b >= Bb) return;
  float mx = -1e30f;
  for (int t = 0; t < T; ++t) mx = fmaxf(mx, scores[t * Bb + b]);
  float sum = 0.f;
  for (int t = 0; t < T; ++t) sum += __expf(scores[t * Bb + b] - mx);
  for (int t = 0; t < T; ++t) att[t * Bb + b] = __expf(scores[t * Bb + b] - mx) / sum;
}

__global__ void att_apply_kernel(const _Float16* __restrict__ hs, const float* __restrict__ att,
                                 _Float16* __restrict__ outp, unsigned Bb, int T, unsigned AC)
{
  unsigned n = blockIdx.x * blockDim.x + threadIdx.x;
  unsigned total = Bb * AC;
  if (n >= total) return;
  unsigned b = n / AC;
  unsigned i = n - b * AC;
  float acc = 0.f;
  for (int t = 0; t < T; ++t)
    acc += att[t * Bb + b] * (float)hs[((long)b * T + t) * AC + i];
  outp[n] = (_Float16)acc;
}

__global__ void concat_ch_kernel(const _Float16* __restrict__ a, const _Float16* __restrict__ b2,
                                 _Float16* __restrict__ o, unsigned N, unsigned Ca, unsigned Cb)
{
  unsigned n = blockIdx.x * blockDim.x + threadIdx.x;
  unsigned Cc = Ca + Cb;
  unsigned total = N * Cc;
  if (n >= total) return;
  unsigned pos = n / Cc;
  unsigned ch = n - pos * Cc;
  o[n] = (ch < Ca) ? a[(long)pos * Ca + ch] : b2[(long)pos * Cb + (ch - Ca)];
}

// ---------------------------------------------------------------------------
static inline void launch_conv(hipStream_t st, const _Float16* in, const _Float16* wt,
                               const float* bias, float* out32, _Float16* out16,
                               int nimg, int Hin, int Win, int Cin,
                               int Hout, int Wout, int Cout,
                               int kh, int kw, int ostride, int ldil, int pad,
                               int iis, int ioff, int ois, int ooff,
                               int flip, int acc, int act)
{
  unsigned M = (unsigned)nimg * (unsigned)(Hout * Wout);
  dim3 grid((M + 15u) / 16u, (unsigned)((Cout + 31) / 32), 1);
  conv_wmma_kernel<<<grid, dim3(32, 1, 1), 0, st>>>(
      in, wt, bias, out32, out16, (unsigned)nimg, Hin, Win, Cin, Hout, Wout, Cout,
      kh, kw, ostride, ldil, pad, iis, ioff, ois, ooff, flip, acc, act);
}

extern "C" void kernel_launch(void* const* d_in, const int* in_sizes, int n_in,
                              void* d_out, int out_size, void* d_ws, size_t ws_size,
                              hipStream_t stream)
{
  (void)in_sizes; (void)n_in; (void)out_size; (void)ws_size;
  auto IN = [&](int i) { return (const float*)d_in[i]; };
  char* wsb = (char*)d_ws;
  float* outp = (float*)d_out;

  const int B = 64, T = 5;
  const int S1 = 61, S2 = 31, S3 = 16, S4 = 8;
  const long A1 = (long)S1 * S1, A2 = (long)S2 * S2, A3 = (long)S3 * S3, A4 = (long)S4 * S4;
  const int C1 = 32, C2 = 64, C3 = 128, C4 = 256;

  // ---- byte-offset workspace carving (256B aligned) ----
  long off = 0;
  auto carve = [&](long elems, int esz) {
    long cur = off;
    off += ((elems * esz + 255) / 256) * 256;
    return cur;
  };
  // f16 activations
  long off_x16 = carve((long)B * T * A1 * 8, 2);
  long off_e1  = carve((long)B * T * A1 * C1, 2);   // enc out / hs scale 1
  long off_e2  = carve((long)B * T * A2 * C2, 2);
  long off_e3  = carve((long)B * T * A3 * C3, 2);
  long off_e4  = carve((long)B * T * A4 * C4, 2);
  long off_at1 = carve((long)B * A1 * C1, 2);
  long off_at2 = carve((long)B * A2 * C2, 2);
  long off_at3 = carve((long)B * A3 * C3, 2);
  long off_at4 = carve((long)B * A4 * C4, 2);
  long off_d4  = carve((long)B * A3 * 128, 2);
  long off_c3  = carve((long)B * A3 * 256, 2);
  long off_d3  = carve((long)B * A2 * 64, 2);
  long off_c2  = carve((long)B * A2 * 128, 2);
  long off_d2  = carve((long)B * A1 * 32, 2);
  long off_c1  = carve((long)B * A1 * 64, 2);
  // f32 buffers
  long off_z   = carve((long)B * A1 * 4 * C1, 4);   // max z (scale 1)
  long off_c   = carve((long)B * A1 * C1, 4);       // max c state
  long off_sc  = carve(512, 4);
  long off_aw  = carve(512, 4);

  auto H16 = [&](long o) { return (_Float16*)(wsb + o); };
  auto F32 = [&](long o) { return (float*)(wsb + o); };

  // ---- f16 weights, transposed to [tap][Cout][Cin] ----
  struct WDef { int idx, taps, Cin, Cout; long woff; };
  WDef wd[16] = {
    { 1, 9,   8,   32, 0}, { 3, 9,  32,   64, 0}, { 5, 4,  64,  128, 0}, { 7, 4, 128,  256, 0},
    { 9, 4,  32,  128, 0}, {10, 4,  32,  128, 0}, {12, 4,  64,  256, 0}, {13, 4,  64,  256, 0},
    {15, 4, 128,  512, 0}, {16, 4, 128,  512, 0}, {18, 4, 256, 1024, 0}, {19, 4, 256, 1024, 0},
    {21, 4, 256,  128, 0}, {23, 4, 256,   64, 0}, {25, 9, 128,   32, 0}, {27, 9,  64,    8, 0},
  };
  for (int i = 0; i < 16; ++i) {
    long ne = (long)wd[i].taps * wd[i].Cin * wd[i].Cout;
    wd[i].woff = carve(ne, 2);
    wcvt_kernel<<<(unsigned)((ne + 255) / 256), 256, 0, stream>>>(
        IN(wd[i].idx), H16(wd[i].woff),
        (unsigned)wd[i].taps, (unsigned)wd[i].Cin, (unsigned)wd[i].Cout);
  }
  auto WT = [&](int idx) -> const _Float16* {
    for (int i = 0; i < 16; ++i) if (wd[i].idx == idx) return H16(wd[i].woff);
    return nullptr;
  };

  // ---- input x -> f16 ----
  {
    unsigned n = (unsigned)((long)B * T * A1 * 8);
    f32_to_f16_kernel<<<(n + 255) / 256, 256, 0, stream>>>(IN(0), H16(off_x16), n);
  }

  // ---- time-distributed encoder (N = B*T images) ----
  launch_conv(stream, H16(off_x16), WT(1), IN(2), nullptr, H16(off_e1), B * T, S1, S1, 8,
              S1, S1, C1, 3, 3, 1, 1, 1, (int)(A1 * 8), 0, (int)(A1 * C1), 0, 0, 0, 1);
  launch_conv(stream, H16(off_e1), WT(3), IN(4), nullptr, H16(off_e2), B * T, S1, S1, C1,
              S2, S2, C2, 3, 3, 2, 1, 1, (int)(A1 * C1), 0, (int)(A2 * C2), 0, 0, 0, 1);
  launch_conv(stream, H16(off_e2), WT(5), IN(6), nullptr, H16(off_e3), B * T, S2, S2, C2,
              S3, S3, C3, 2, 2, 2, 1, 0, (int)(A2 * C2), 0, (int)(A3 * C3), 0, 0, 0, 1);
  launch_conv(stream, H16(off_e3), WT(7), IN(8), nullptr, H16(off_e4), B * T, S3, S3, C3,
              S4, S4, C4, 2, 2, 2, 1, 0, (int)(A3 * C3), 0, (int)(A4 * C4), 0, 0, 0, 1);

  // ---- ConvLSTM + temporal attention per scale ----
  // h_t overwrites e[:,t] in place (conv_x of step t consumes e[:,t] first).
  struct ScaleP { int S; long A; int C; long eoff; long atoff; int iwx, iwh, ib; };
  ScaleP scs[4] = {
    { S1, A1, C1, off_e1, off_at1,  9, 10, 11 },
    { S2, A2, C2, off_e2, off_at2, 12, 13, 14 },
    { S3, A3, C3, off_e3, off_at3, 15, 16, 17 },
    { S4, A4, C4, off_e4, off_at4, 18, 19, 20 },
  };
  for (int s = 0; s < 4; ++s) {
    ScaleP& P = scs[s];
    _Float16* ebuf = H16(P.eoff);
    float* z = F32(off_z);
    float* cst = F32(off_c);
    int AC = (int)(P.A * P.C);
    for (int t = 0; t < T; ++t) {
      // z = conv(x_t, wx) + b   (2x2 SAME, stride 1 -> pad_lo 0)
      launch_conv(stream, ebuf, WT(P.iwx), IN(P.ib), z, nullptr, B, P.S, P.S, P.C,
                  P.S, P.S, 4 * P.C, 2, 2, 1, 1, 0,
                  T * AC, t * AC, (int)(P.A * 4 * P.C), 0, 0, 0, 0);
      // z += conv(h_{t-1}, wh)
      if (t > 0)
        launch_conv(stream, ebuf, WT(P.iwh), nullptr, z, nullptr, B, P.S, P.S, P.C,
                    P.S, P.S, 4 * P.C, 2, 2, 1, 1, 0,
                    T * AC, (t - 1) * AC, (int)(P.A * 4 * P.C), 0, 0, 1, 0);
      unsigned tot = (unsigned)((long)B * AC);
      lstm_gates_kernel<<<(tot + 255) / 256, 256, 0, stream>>>(
          z, cst, ebuf, (unsigned)B, (unsigned)P.A, (unsigned)P.C, t, T, (t == 0) ? 1 : 0);
    }
    att_scores_kernel<<<T * B, 256, 0, stream>>>(ebuf, F32(off_sc), B, T, AC);
    att_softmax_kernel<<<1, 64, 0, stream>>>(F32(off_sc), F32(off_aw), B, T);
    unsigned tot = (unsigned)((long)B * AC);
    att_apply_kernel<<<(tot + 255) / 256, 256, 0, stream>>>(
        ebuf, F32(off_aw), H16(P.atoff), (unsigned)B, T, (unsigned)AC);
  }

  // ---- decoder (transposed convs = lhs-dilated conv with flipped kernel) ----
  // d4 = selu(tconv(att4, W4d, stride 2, pad (1,1))) : 8 -> 16, 256 -> 128
  launch_conv(stream, H16(off_at4), WT(21), IN(22), nullptr, H16(off_d4), B, S4, S4, 256,
              S3, S3, 128, 2, 2, 1, 2, 1, (int)(A4 * 256), 0, (int)(A3 * 128), 0, 1, 0, 1);
  {
    unsigned N = (unsigned)((long)B * A3), tot = N * 256;
    concat_ch_kernel<<<(tot + 255) / 256, 256, 0, stream>>>(
        H16(off_d4), H16(off_at3), H16(off_c3), N, 128, 128);
  }
  // d3 = selu(tconv(cat3, W3d, stride 2, pad (1,0))) : 16 -> 31, 256 -> 64
  launch_conv(stream, H16(off_c3), WT(23), IN(24), nullptr, H16(off_d3), B, S3, S3, 256,
              S2, S2, 64, 2, 2, 1, 2, 1, (int)(A3 * 256), 0, (int)(A2 * 64), 0, 1, 0, 1);
  {
    unsigned N = (unsigned)((long)B * A2), tot = N * 128;
    concat_ch_kernel<<<(tot + 255) / 256, 256, 0, stream>>>(
        H16(off_d3), H16(off_at2), H16(off_c2), N, 64, 64);
  }
  // d2 = selu(tconv(cat2, W2d 3x3, stride 2, pad (1,1))) : 31 -> 61, 128 -> 32
  launch_conv(stream, H16(off_c2), WT(25), IN(26), nullptr, H16(off_d2), B, S2, S2, 128,
              S1, S1, 32, 3, 3, 1, 2, 1, (int)(A2 * 128), 0, (int)(A1 * 32), 0, 1, 0, 1);
  {
    unsigned N = (unsigned)((long)B * A1), tot = N * 64;
    concat_ch_kernel<<<(tot + 255) / 256, 256, 0, stream>>>(
        H16(off_d2), H16(off_at1), H16(off_c1), N, 32, 32);
  }
  // out = selu(tconv(cat1, W1d 3x3, stride 1, pad (1,1))) : 61 -> 61, 64 -> 8 (f32 out)
  launch_conv(stream, H16(off_c1), WT(27), IN(28), outp, nullptr, B, S1, S1, 64,
              S1, S1, 8, 3, 3, 1, 1, 1, (int)(A1 * 64), 0, (int)(A1 * 8), 0, 1, 0, 1);
}
